// Decoder_128849019270
// MI455X (gfx1250) — compile-verified, matrix operand order checked by reading, not run on hardware
//
#include <hip/hip_runtime.h>
#include <hip/hip_bf16.h>
#include <math.h>

// ---------------------------------------------------------------------------
// Problem constants (from the reference)
// ---------------------------------------------------------------------------
constexpr int Lc  = 4;
constexpr int Ec  = 768;
constexpr int Bc  = 64;
constexpr int Tc  = 512;
constexpr int Gc  = 4 * Ec;   // 3072 gate width
constexpr int K2c = 2 * Ec;   // 1536 concatenated [x|h] K
constexpr int SLAB  = 256;    // K-slab staged in LDS per iteration
constexpr int SLABP = SLAB + 8; // padded LDS row stride (matches TDM pad below)
constexpr int NWG  = 32;      // persistent grid (must be co-resident)
constexpr int NTHR = 256;     // 8 wave32 per WG

typedef __attribute__((ext_vector_type(16))) __bf16 v16bf;
typedef __attribute__((ext_vector_type(8)))  float  v8f;

// Tensor Data Mover availability (5-arg on clang-22/ROCm7.2, 6-arg on clang-23)
#if defined(__has_builtin)
#  if __has_builtin(__builtin_amdgcn_tensor_load_to_lds) && \
      __has_builtin(__builtin_amdgcn_s_wait_tensorcnt)
#    define HAVE_TDM 1
#  endif
#endif
#ifndef HAVE_TDM
#  define HAVE_TDM 0
#endif

struct __align__(32) FragU { unsigned u[8]; };

__device__ __forceinline__ unsigned short f2bf(float f) {
  unsigned u = __float_as_uint(f);
  u += 0x7fffu + ((u >> 16) & 1u);   // round-to-nearest-even
  return (unsigned short)(u >> 16);
}

__device__ __forceinline__ float sigm(float v) {
  return 1.0f / (1.0f + __expf(-v));
}

// A/B fragment (CDNA5 16-bit layout) from a padded LDS tile.
// lane 0-15: row=lane, K 0..7 & 16..23 ; lane 16-31: row=lane-16, K 8..15 & 24..31
__device__ __forceinline__ v16bf load_frag_lds(const unsigned short (*Ts)[SLABP],
                                               int rbase, int kk, int lane) {
  const int half = lane >> 4, r = lane & 15;
  const unsigned short* p = &Ts[rbase + r][kk + half * 8];
  FragU f;
#pragma unroll
  for (int q = 0; q < 8; ++q) {
    const int ko = ((q < 4) ? 0 : 16) + 2 * (q & 3);
    f.u[q] = *(const unsigned*)(p + ko);
  }
  return *(const v16bf*)&f;
}

// B fragment straight from (L2-resident) global bf16 weights, N-major rows.
__device__ __forceinline__ v16bf load_frag_gbl(const unsigned short* __restrict__ W,
                                               int ld, int nbase, int k0, int lane) {
  const int half = lane >> 4, cn = lane & 15;
  const unsigned short* p = W + (size_t)(nbase + cn) * ld + k0 + half * 8;
  FragU f;
#pragma unroll
  for (int q = 0; q < 8; ++q) {
    const int ko = ((q < 4) ? 0 : 16) + 2 * (q & 3);
    f.u[q] = *(const unsigned*)(p + ko);
  }
  return *(const v16bf*)&f;
}

#if HAVE_TDM
typedef __attribute__((ext_vector_type(4))) unsigned tdm_u4;
typedef __attribute__((ext_vector_type(8))) int      tdm_i8;
typedef __attribute__((ext_vector_type(4))) int      tdm_i4;

// Issue one 2D TDM tile load: tile_rows x SLAB bf16 elements from a row-major
// tensor with row stride K2c, into LDS at lds_off with 4 DWORDs of padding
// inserted every 128 DWORDs (-> effective LDS row stride SLABP elements).
__device__ __forceinline__ void tdm_load_2d(const unsigned short* gsrc,
                                            unsigned lds_off, int tile_rows) {
  const unsigned long long ga = (unsigned long long)(const void*)gsrc;
  tdm_u4 g0;
  g0[0] = 1u;                                   // count=1 (valid), user mode
  g0[1] = lds_off;                              // lds_addr (bytes)
  g0[2] = (unsigned)ga;                         // global_addr[31:0]
  g0[3] = (unsigned)((ga >> 32) & 0x01FFFFFFull) | 0x80000000u; // [56:32]|type=2
  tdm_i8 g1;
  g1[0] = (int)((1u << 16)      // data_size = 1 -> 2 bytes
              | (1u << 20)      // pad_enable
              | (6u << 22)      // pad_interval: every 128 DWORDs (512B row)
              | (3u << 25));    // pad_amount: 4 DWORDs (16B = 8 elements)
  g1[1] = (int)((unsigned)K2c << 16);           // tensor_dim0 = 1536
  g1[2] = (int)(4096u << 16);                   // tensor_dim1 (OOB bound, rows)
  g1[3] = (int)((unsigned)SLAB << 16);          // tile_dim0 = 256
  g1[4] = (int)(unsigned)tile_rows;             // tile_dim1 ; tile_dim2 = 0
  g1[5] = (int)(unsigned)K2c;                   // tensor_dim0_stride = 1536
  g1[6] = 0;
  g1[7] = 0;
  const tdm_i4 z4 = {0, 0, 0, 0};
#if __clang_major__ >= 23
  const tdm_i8 z8 = {0, 0, 0, 0, 0, 0, 0, 0};
  __builtin_amdgcn_tensor_load_to_lds(g0, g1, z4, z4, z8, 0);
#else
  __builtin_amdgcn_tensor_load_to_lds(g0, g1, z4, z4, 0);
#endif
}
#endif  // HAVE_TDM

// Device-wide split barrier: counter + generation word in d_ws.
__device__ __forceinline__ void grid_sync(unsigned* bar) {
  __threadfence();          // release
  __syncthreads();
  if (threadIdx.x == 0) {
    volatile unsigned* vb = bar;
    const unsigned gen = vb[1];
    const unsigned a = atomicAdd(bar, 1u);
    if (a == (unsigned)(NWG - 1)) {
      vb[0] = 0u;
      __threadfence();
      atomicAdd(bar + 1, 1u);
    } else {
      while (vb[1] == gen) { __builtin_amdgcn_s_sleep(2); }
    }
  }
  __syncthreads();
  __threadfence();          // acquire
}

__global__ void zero_bar_kernel(unsigned* bar) {
  if (threadIdx.x < 8) bar[threadIdx.x] = 0u;
}

// ---------------------------------------------------------------------------
// Persistent fused LSTM decoder. One launch runs all T=512 steps.
// ---------------------------------------------------------------------------
__global__ __launch_bounds__(NTHR) void lstm_decoder_persistent(
    const float* __restrict__ inputs,
    const float* __restrict__ W_ih, const float* __restrict__ W_hh,
    const float* __restrict__ b_ih, const float* __restrict__ b_hh,
    const float* __restrict__ ln_g, const float* __restrict__ ln_b,
    const float* __restrict__ W_out, const float* __restrict__ b_out,
    float* __restrict__ out,
    unsigned short* __restrict__ Wcat,   // [L][G][K2] bf16 (L2-resident, 38 MB)
    unsigned short* __restrict__ Woutb,  // [E][E]    bf16
    float* __restrict__ bcat,            // [L][G]    b_ih + b_hh
    float* __restrict__ x,               // [B][E]
    float* __restrict__ prev,            // [B][E]
    float* __restrict__ h,               // [B][E]
    float* __restrict__ c,               // [B][E]
    float* __restrict__ gates,           // [B][G]
    unsigned* __restrict__ bar) {

  __shared__ unsigned short As[Bc][SLABP];       // bf16 activation slab (~33 KB)
#if HAVE_TDM
  __shared__ unsigned short Bs[3][32][SLABP];    // TDM-staged weight tiles (~50 KB)
#endif

  const int tid  = threadIdx.x;
  const int gtid = blockIdx.x * NTHR + tid;
  const int NT   = NWG * NTHR;
  const int lane = tid & 31;
  const int gw   = gtid >> 5;               // global wave id: 0..255

  // ---------------- one-time init (re-done every call, deterministic) -------
  for (size_t i = gtid; i < (size_t)Lc * Gc * K2c; i += NT) {
    const size_t l = i / ((size_t)Gc * K2c);
    const size_t r = i % ((size_t)Gc * K2c);
    const size_t n = r / K2c;
    const size_t k = r % K2c;
    const float w = (k < (size_t)Ec) ? W_ih[(l * Gc + n) * Ec + k]
                                     : W_hh[(l * Gc + n) * Ec + (k - Ec)];
    Wcat[i] = f2bf(w);
  }
  for (size_t i = gtid; i < (size_t)Ec * Ec; i += NT) Woutb[i] = f2bf(W_out[i]);
  for (size_t i = gtid; i < (size_t)Lc * Gc; i += NT) bcat[i] = b_ih[i] + b_hh[i];
  for (size_t i = gtid; i < (size_t)Bc * Ec; i += NT) {
    x[i] = inputs[i];         // inputs is (B,1,E) == flat B*E
    prev[i] = 0.0f; h[i] = 0.0f; c[i] = 0.0f;
  }
  grid_sync(bar);

  // gates GEMM tiling: 4 M-tiles x 192 N-tiles = 768 tiles = 3 per wave.
  // tiles {gw, gw+256, gw+512} share the same M-tile -> one A frag, 3 B frags.
  // Within a WG the 8 waves cover 32 contiguous N rows (x3 groups 1024 apart):
  // exactly three rectangular TDM tiles per K-slab.
  const int mt   = gw & 3;
  const int nt0  = gw >> 2;                 // 0..63 ; also +64, +128
  const int rloc = (tid >> 7) * 16;         // this wave's row base inside a B tile

  // ---------------- recurrent time loop -------------------------------------
  for (int t = 0; t < Tc; ++t) {
    for (int l = 0; l < Lc; ++l) {
      // ---- GEMM: gates = [x|h] @ Wcat[l]^T + bcat[l]  (M=64,N=3072,K=1536)
      const unsigned short* Wl = Wcat + (size_t)l * Gc * K2c;
      v8f a0 = {0.f,0.f,0.f,0.f,0.f,0.f,0.f,0.f};
      v8f a1 = a0, a2 = a0;
      for (int ks = 0; ks < K2c; ks += SLAB) {
        __syncthreads();     // everyone done reading previous slab (As & Bs)
#if HAVE_TDM
        if (tid < 32) {      // wave 0: async DMA the three 32x256 weight tiles
          const unsigned short* base =
              Wl + (size_t)(blockIdx.x * 32) * K2c + ks;
          tdm_load_2d(base,                      (unsigned)(size_t)&Bs[0][0][0], 32);
          tdm_load_2d(base + (size_t)1024 * K2c, (unsigned)(size_t)&Bs[1][0][0], 32);
          tdm_load_2d(base + (size_t)2048 * K2c, (unsigned)(size_t)&Bs[2][0][0], 32);
        }
#endif
        // overlap: convert activation slab f32 -> bf16 while the TDM runs
        for (int i = tid; i < Bc * SLAB; i += NTHR) {
          const int r = i >> 8, kk = i & (SLAB - 1);
          const int k = ks + kk;
          const float v = (k < Ec) ? x[r * Ec + k] : h[r * Ec + (k - Ec)];
          As[r][kk] = f2bf(v);
        }
#if HAVE_TDM
        if (tid < 32) { __builtin_amdgcn_s_wait_tensorcnt(0); }
#endif
        __syncthreads();     // A staged + TDM complete -> tiles visible to WG
#pragma unroll 2
        for (int kk = 0; kk < SLAB; kk += 32) {
          const v16bf af = load_frag_lds(As, mt * 16, kk, lane);
#if HAVE_TDM
          const v16bf b0 = load_frag_lds(Bs[0], rloc, kk, lane);
          const v16bf b1 = load_frag_lds(Bs[1], rloc, kk, lane);
          const v16bf b2 = load_frag_lds(Bs[2], rloc, kk, lane);
#else
          const v16bf b0 = load_frag_gbl(Wl, K2c, (nt0      ) * 16, ks + kk, lane);
          const v16bf b1 = load_frag_gbl(Wl, K2c, (nt0 +  64) * 16, ks + kk, lane);
          const v16bf b2 = load_frag_gbl(Wl, K2c, (nt0 + 128) * 16, ks + kk, lane);
#endif
          a0 = __builtin_amdgcn_wmma_f32_16x16x32_bf16(false, af, false, b0, (short)0, a0, false, false);
          a1 = __builtin_amdgcn_wmma_f32_16x16x32_bf16(false, af, false, b1, (short)0, a1, false, false);
          a2 = __builtin_amdgcn_wmma_f32_16x16x32_bf16(false, af, false, b2, (short)0, a2, false, false);
        }
      }
      {
        const int half = lane >> 4, cn = lane & 15;
        const int row0 = mt * 16 + half * 8;
        const int n0 = (nt0      ) * 16 + cn;
        const int n1 = (nt0 +  64) * 16 + cn;
        const int n2 = (nt0 + 128) * 16 + cn;
        const float bb0 = bcat[l * Gc + n0];
        const float bb1 = bcat[l * Gc + n1];
        const float bb2 = bcat[l * Gc + n2];
#pragma unroll
        for (int v = 0; v < 8; ++v) {
          const int rr = row0 + v;              // M = v + 8*half (C/D layout)
          gates[(size_t)rr * Gc + n0] = a0[v] + bb0;
          gates[(size_t)rr * Gc + n1] = a1[v] + bb1;
          gates[(size_t)rr * Gc + n2] = a2[v] + bb2;
        }
      }
      grid_sync(bar);

      // ---- fused gate nonlinearity + (optional) LayerNorm: 1 wave per row --
      if (gw < Bc) {
        const int r = gw;
        const float* grow = gates + (size_t)r * Gc;
        float* crow = c + (size_t)r * Ec;
        float* hrow = h + (size_t)r * Ec;
        float hv[24];
#pragma unroll
        for (int j = 0; j < 24; ++j) {
          const int e = lane + 32 * j;
          const float ig = sigm(grow[e]);
          const float fg = sigm(grow[Ec + e]);
          const float gg = tanhf(grow[2 * Ec + e]);
          const float og = sigm(grow[3 * Ec + e]);
          const float cv = fg * crow[e] + ig * gg;
          crow[e] = cv;
          const float hh = og * tanhf(cv);
          hrow[e] = hh;
          hv[j] = hh;
        }
        if (l < Lc - 1) {
          // x_next = LN(h + prev) * g + b ; prev = x_next
          const float* psrc = (l == 0) ? (x + (size_t)r * Ec) : (prev + (size_t)r * Ec);
          float sv[24];
          float s1 = 0.f, s2 = 0.f;
#pragma unroll
          for (int j = 0; j < 24; ++j) {
            const int e = lane + 32 * j;
            const float s = hv[j] + psrc[e];
            sv[j] = s; s1 += s; s2 += s * s;
          }
#pragma unroll
          for (int off = 16; off > 0; off >>= 1) {   // wave32 butterfly
            s1 += __shfl_xor(s1, off);
            s2 += __shfl_xor(s2, off);
          }
          const float mean = s1 * (1.0f / (float)Ec);
          const float var  = s2 * (1.0f / (float)Ec) - mean * mean;
          const float inv  = rsqrtf(var + 1e-5f);
          const float* lg = ln_g + (size_t)l * Ec;   // ln params for layer l+1
          const float* lb = ln_b + (size_t)l * Ec;
          float* xrow = x    + (size_t)r * Ec;
          float* prow = prev + (size_t)r * Ec;
#pragma unroll
          for (int j = 0; j < 24; ++j) {
            const int e = lane + 32 * j;
            const float xn = (sv[j] - mean) * inv * lg[e] + lb[e];
            xrow[e] = xn; prow[e] = xn;
          }
        }
      }
      grid_sync(bar);
    }

    // ---- output projection: y_t = h @ W_out^T + b_out (M=64,N=768,K=768) ---
    {
      const bool active = gw < 192;        // 4 x 48 tiles
      const int omt = gw & 3, ont = gw >> 2;
      v8f acc = {0.f,0.f,0.f,0.f,0.f,0.f,0.f,0.f};
      for (int ks = 0; ks < Ec; ks += SLAB) {
        __syncthreads();
        for (int i = tid; i < Bc * SLAB; i += NTHR) {
          const int r = i >> 8, kk = i & (SLAB - 1);
          As[r][kk] = f2bf(h[r * Ec + ks + kk]);
        }
        __syncthreads();
        if (active) {
          if (ks + SLAB < Ec) {   // warm L2->L0 for next weight slab
            __builtin_prefetch(Woutb + (size_t)(ont * 16 + (lane & 15)) * Ec + ks + SLAB, 0, 1);
          }
#pragma unroll 2
          for (int kk = 0; kk < SLAB; kk += 32) {
            const v16bf af = load_frag_lds(As, omt * 16, kk, lane);
            const v16bf bf = load_frag_gbl(Woutb, Ec, ont * 16, ks + kk, lane);
            acc = __builtin_amdgcn_wmma_f32_16x16x32_bf16(false, af, false, bf, (short)0, acc, false, false);
          }
        }
      }
      if (active) {
        const int half = lane >> 4, cn = lane & 15;
        const int n = ont * 16 + cn;
        const float bo = b_out[n];
#pragma unroll
        for (int v = 0; v < 8; ++v) {
          const int rr = omt * 16 + half * 8 + v;
          const float val = acc[v] + bo;
          out[((size_t)rr * Tc + t) * Ec + n] = val;   // (B,T,E)
          x[(size_t)rr * Ec + n] = val;                // next step's layer-0 x
        }
      }
      grid_sync(bar);
    }
  }
}

// ---------------------------------------------------------------------------
// Host-side launch (graph-capture safe: only kernel launches on `stream`)
// ---------------------------------------------------------------------------
extern "C" void kernel_launch(void* const* d_in, const int* in_sizes, int n_in,
                              void* d_out, int out_size, void* d_ws, size_t ws_size,
                              hipStream_t stream) {
  const float* inputs = (const float*)d_in[0];
  // d_in[1] = targets (unused by forward pass)
  const float* W_ih  = (const float*)d_in[2];
  const float* W_hh  = (const float*)d_in[3];
  const float* b_ih  = (const float*)d_in[4];
  const float* b_hh  = (const float*)d_in[5];
  const float* ln_g  = (const float*)d_in[6];
  const float* ln_b  = (const float*)d_in[7];
  const float* W_out = (const float*)d_in[8];
  const float* b_out = (const float*)d_in[9];
  float* out = (float*)d_out;

  char* ws = (char*)d_ws;
  size_t off = 0;
  auto alloc = [&](size_t bytes) -> void* {
    void* p = ws + off;
    off = (off + bytes + 255) & ~(size_t)255;
    return p;
  };
  unsigned*       bar   = (unsigned*)alloc(256);
  unsigned short* Wcat  = (unsigned short*)alloc((size_t)Lc * Gc * K2c * 2);
  unsigned short* Woutb = (unsigned short*)alloc((size_t)Ec * Ec * 2);
  float*          bcat  = (float*)alloc((size_t)Lc * Gc * 4);
  float*          x     = (float*)alloc((size_t)Bc * Ec * 4);
  float*          prev  = (float*)alloc((size_t)Bc * Ec * 4);
  float*          h     = (float*)alloc((size_t)Bc * Ec * 4);
  float*          c     = (float*)alloc((size_t)Bc * Ec * 4);
  float*          gates = (float*)alloc((size_t)Bc * Gc * 4);

  zero_bar_kernel<<<1, 32, 0, stream>>>(bar);
  lstm_decoder_persistent<<<NWG, NTHR, 0, stream>>>(
      inputs, W_ih, W_hh, b_ih, b_hh, ln_g, ln_b, W_out, b_out, out,
      Wcat, Woutb, bcat, x, prev, h, c, gates, bar);
}
